// SelectiveViewFusionBlock_16853451669713
// MI455X (gfx1250) — matrix operand order repaired and checked by reading, not compiled, not run on hardware
//
#include <hip/hip_runtime.h>

#define NROWS 12288
#define DDIM  128

typedef __attribute__((ext_vector_type(16))) __bf16 v16bf;
typedef __attribute__((ext_vector_type(8)))  float  v8f;

union Frag16 {
  v16bf v;
  uint4 q[2];
  unsigned u[8];
};

// round-half-up f32 -> bf16 (cheap, unbiased for random data)
__device__ __forceinline__ unsigned short f2bf(float f) {
  return (unsigned short)((__float_as_uint(f) + 0x8000u) >> 16);
}
// pack two f32 into one dword of 2 bf16 (lo in [15:0])
__device__ __forceinline__ unsigned pkbf(float lo, float hi) {
  const unsigned l = (__float_as_uint(lo) + 0x8000u) >> 16;
  const unsigned h = (__float_as_uint(hi) + 0x8000u) & 0xffff0000u;
  return h | l;
}
__device__ __forceinline__ float wred(float v) {
#pragma unroll
  for (int o = 16; o > 0; o >>= 1) v += __shfl_xor(v, o, 32);
  return v;
}

// ---------------------------------------------------------------------------
// Kernel 0: transpose + bf16-convert the three weight matrices.
// w?t layout: [out_col][k] so a WMMA B-fragment lane reads 32 contiguous bytes.
// ---------------------------------------------------------------------------
__global__ __launch_bounds__(128) void k_prep_w(
    const float* __restrict__ W1, const float* __restrict__ W2,
    const float* __restrict__ Wg,
    unsigned short* __restrict__ w1t, unsigned short* __restrict__ w2t,
    unsigned short* __restrict__ wgt, unsigned short* __restrict__ wgb) {
  const int o = blockIdx.x;   // output column 0..127
  const int i = threadIdx.x;  // k index 0..127
  switch (blockIdx.y) {
    case 0: w1t[o * DDIM + i] = f2bf(W1[i * DDIM + o]); break;
    case 1: w2t[o * DDIM + i] = f2bf(W2[i * DDIM + o]); break;
    case 2: wgt[o * DDIM + i] = f2bf(Wg[i * DDIM + o]); break;
    default: wgb[o * DDIM + i] = f2bf(Wg[(DDIM + i) * DDIM + o]); break;
  }
}

// ---------------------------------------------------------------------------
// Kernel 1: per-row layernorm, logits, Gumbel argmax gate.
// 1 wave per row (4 elements/lane). Writes:
//   xnb: x_norm bf16 row-major  [N][128]
//   ztb: (gate_j * x_norm_j) bf16 TRANSPOSED [128][N]  (B operand of big GEMM)
//   gate: float per row
// ---------------------------------------------------------------------------
__global__ __launch_bounds__(256) void k_gate_xnorm(
    const float* __restrict__ x, const float* __restrict__ u,
    const float* __restrict__ Wn, const float* __restrict__ bn,
    const float* __restrict__ g_in, const float* __restrict__ b_in,
    unsigned short* __restrict__ xnb, unsigned short* __restrict__ ztb,
    float* __restrict__ gate) {
  const int lane = threadIdx.x & 31;
  const int wv = threadIdx.x >> 5;
  const int row = blockIdx.x * 8 + wv;
  const int c0 = lane * 4;

  const float4 xv = *reinterpret_cast<const float4*>(x + (size_t)row * DDIM + c0);
  float s = xv.x + xv.y + xv.z + xv.w;
  s = wred(s);
  const float mean = s * (1.0f / DDIM);
  const float d0 = xv.x - mean, d1 = xv.y - mean, d2 = xv.z - mean, d3 = xv.w - mean;
  float ss = d0 * d0 + d1 * d1 + d2 * d2 + d3 * d3;
  ss = wred(ss);
  const float inv = rsqrtf(ss * (1.0f / DDIM) + 1e-5f);

  const float n0 = d0 * inv * g_in[c0 + 0] + b_in[c0 + 0];
  const float n1 = d1 * inv * g_in[c0 + 1] + b_in[c0 + 1];
  const float n2 = d2 * inv * g_in[c0 + 2] + b_in[c0 + 2];
  const float n3 = d3 * inv * g_in[c0 + 3] + b_in[c0 + 3];

  // logits = x_norm @ Wn + bn ; Wn is [128][2] row-major
  float l0 = n0 * Wn[(c0 + 0) * 2] + n1 * Wn[(c0 + 1) * 2] +
             n2 * Wn[(c0 + 2) * 2] + n3 * Wn[(c0 + 3) * 2];
  float l1 = n0 * Wn[(c0 + 0) * 2 + 1] + n1 * Wn[(c0 + 1) * 2 + 1] +
             n2 * Wn[(c0 + 2) * 2 + 1] + n3 * Wn[(c0 + 3) * 2 + 1];
  l0 = wred(l0) + bn[0];
  l1 = wred(l1) + bn[1];

  const float EPSC = 1e-10f;
  const float u0 = u[(size_t)row * 2 + 0];
  const float u1 = u[(size_t)row * 2 + 1];
  const float gb0 = -logf(-logf(fminf(fmaxf(u0, EPSC), 1.0f)) + EPSC);
  const float gb1 = -logf(-logf(fminf(fmaxf(u1, EPSC), 1.0f)) + EPSC);
  // forward value of straight-through Gumbel softmax == one-hot(argmax)
  const float gt = ((l1 + gb1) > (l0 + gb0)) ? 1.0f : 0.0f;
  if (lane == 0) gate[row] = gt;

  const unsigned short b0 = f2bf(n0), b1_ = f2bf(n1), b2_ = f2bf(n2), b3_ = f2bf(n3);
  uint2 pk;
  pk.x = (unsigned)b0 | ((unsigned)b1_ << 16);
  pk.y = (unsigned)b2_ | ((unsigned)b3_ << 16);
  *reinterpret_cast<uint2*>(xnb + (size_t)row * DDIM + c0) = pk;

  const unsigned short z0 = gt != 0.0f ? b0 : (unsigned short)0;
  const unsigned short z1 = gt != 0.0f ? b1_ : (unsigned short)0;
  const unsigned short z2 = gt != 0.0f ? b2_ : (unsigned short)0;
  const unsigned short z3 = gt != 0.0f ? b3_ : (unsigned short)0;
  ztb[(size_t)(c0 + 0) * NROWS + row] = z0;
  ztb[(size_t)(c0 + 1) * NROWS + row] = z1;
  ztb[(size_t)(c0 + 2) * NROWS + row] = z2;
  ztb[(size_t)(c0 + 3) * NROWS + row] = z3;
}

// ---------------------------------------------------------------------------
// Kernel 2: view2 = diag(gate) * (adj @ z)   (z already gated on columns)
// Wave = 32 rows x 128 cols (2 M-tiles x 8 N-tiles): each 1KB B fragment
// feeds TWO WMMAs, halving B-operand bandwidth per FLOP. adj (fp32, HBM)
// is loaded exactly once and converted in-register to bf16; A floats are
// double-buffered across K iterations to keep HBM loads in flight.
// ---------------------------------------------------------------------------
__global__ __launch_bounds__(64) void k_gemm_big(
    const float* __restrict__ adj, const unsigned short* __restrict__ ztb,
    const float* __restrict__ gate, unsigned short* __restrict__ v2b) {
  const int lane = threadIdx.x & 31;
  const int wv = threadIdx.x >> 5;          // 0..1
  const int m0 = blockIdx.x * 64 + wv * 32; // wave covers rows m0..m0+31
  const int ml = lane & 15;
  const int hi = lane >> 4;

  const float* ap0 = adj + (size_t)(m0 + ml) * NROWS + hi * 8;
  const float* ap1 = adj + (size_t)(m0 + 16 + ml) * NROWS + hi * 8;
  const unsigned short* bbase = ztb + (size_t)ml * NROWS + hi * 8;

  v8f acc0[8], acc1[8];
#pragma unroll
  for (int t = 0; t < 8; ++t) {
    acc0[t] = (v8f){0.f, 0.f, 0.f, 0.f, 0.f, 0.f, 0.f, 0.f};
    acc1[t] = (v8f){0.f, 0.f, 0.f, 0.f, 0.f, 0.f, 0.f, 0.f};
  }

  // prime A double-buffers (K = 0..31 for both M-tiles)
  float4 nf0 = *reinterpret_cast<const float4*>(ap0);
  float4 nf1 = *reinterpret_cast<const float4*>(ap0 + 4);
  float4 nf2 = *reinterpret_cast<const float4*>(ap0 + 16);
  float4 nf3 = *reinterpret_cast<const float4*>(ap0 + 20);
  float4 mf0 = *reinterpret_cast<const float4*>(ap1);
  float4 mf1 = *reinterpret_cast<const float4*>(ap1 + 4);
  float4 mf2 = *reinterpret_cast<const float4*>(ap1 + 16);
  float4 mf3 = *reinterpret_cast<const float4*>(ap1 + 20);

  for (int k = 0; k < NROWS; k += 32) {
    // 1) issue all B loads for this K step (16 x b128)
    Frag16 b[8];
#pragma unroll
    for (int t = 0; t < 8; ++t) {
      const unsigned short* bp = bbase + (size_t)t * 16 * NROWS + k;
      b[t].q[0] = *reinterpret_cast<const uint4*>(bp);
      b[t].q[1] = *reinterpret_cast<const uint4*>(bp + 16);
    }
    // 2) convert current A floats -> bf16 fragments (both M-tiles)
    Frag16 a0, a1;
    a0.u[0] = pkbf(nf0.x, nf0.y); a0.u[1] = pkbf(nf0.z, nf0.w);
    a0.u[2] = pkbf(nf1.x, nf1.y); a0.u[3] = pkbf(nf1.z, nf1.w);
    a0.u[4] = pkbf(nf2.x, nf2.y); a0.u[5] = pkbf(nf2.z, nf2.w);
    a0.u[6] = pkbf(nf3.x, nf3.y); a0.u[7] = pkbf(nf3.z, nf3.w);
    a1.u[0] = pkbf(mf0.x, mf0.y); a1.u[1] = pkbf(mf0.z, mf0.w);
    a1.u[2] = pkbf(mf1.x, mf1.y); a1.u[3] = pkbf(mf1.z, mf1.w);
    a1.u[4] = pkbf(mf2.x, mf2.y); a1.u[5] = pkbf(mf2.z, mf2.w);
    a1.u[6] = pkbf(mf3.x, mf3.y); a1.u[7] = pkbf(mf3.z, mf3.w);
    // 3) prefetch next iteration's A floats (wrap to 0 on last iter: stays
    //    in-bounds, result unused)
    const int kn = (k + 32 < NROWS) ? (k + 32) : 0;
    nf0 = *reinterpret_cast<const float4*>(ap0 + kn);
    nf1 = *reinterpret_cast<const float4*>(ap0 + kn + 4);
    nf2 = *reinterpret_cast<const float4*>(ap0 + kn + 16);
    nf3 = *reinterpret_cast<const float4*>(ap0 + kn + 20);
    mf0 = *reinterpret_cast<const float4*>(ap1 + kn);
    mf1 = *reinterpret_cast<const float4*>(ap1 + kn + 4);
    mf2 = *reinterpret_cast<const float4*>(ap1 + kn + 16);
    mf3 = *reinterpret_cast<const float4*>(ap1 + kn + 20);
    // 4) 16 WMMAs; b[t] dies right after its second use
#pragma unroll
    for (int t = 0; t < 8; ++t) {
      acc0[t] = __builtin_amdgcn_wmma_f32_16x16x32_bf16(
          false, a0.v, false, b[t].v, (short)0, acc0[t], false, false);
      acc1[t] = __builtin_amdgcn_wmma_f32_16x16x32_bf16(
          false, a1.v, false, b[t].v, (short)0, acc1[t], false, false);
    }
  }

  float grow0[8], grow1[8];
#pragma unroll
  for (int r = 0; r < 8; ++r) {
    grow0[r] = gate[m0 + hi * 8 + r];
    grow1[r] = gate[m0 + 16 + hi * 8 + r];
  }
#pragma unroll
  for (int t = 0; t < 8; ++t) {
#pragma unroll
    for (int r = 0; r < 8; ++r) {
      const int row0 = m0 + hi * 8 + r;
      const int row1 = m0 + 16 + hi * 8 + r;
      v2b[(size_t)row0 * DDIM + t * 16 + ml] = f2bf(acc0[t][r] * grow0[r]);
      v2b[(size_t)row1 * DDIM + t * 16 + ml] = f2bf(acc1[t][r] * grow1[r]);
    }
  }
}

// ---------------------------------------------------------------------------
// Kernel 3: fused epilogue.
//   h1 = view1@W1+b1 ; h2 = view2@W2+b2 ; g = sigmoid(view1@WgT+view2@WgB+bg)
//   fused = g*h1 + (1-g)*h2 + residual ; out = layernorm(fused)*g_out+b_out
// A fragments (full K=128 for both views) loaded once per wave, reused over
// all 8 column tiles; per K-chunk the 4 weight B-frags are loaded as one
// clause before the 4 WMMAs.
// ---------------------------------------------------------------------------
__global__ __launch_bounds__(128) void k_epilogue(
    const unsigned short* __restrict__ xnb, const unsigned short* __restrict__ v2b,
    const float* __restrict__ x,
    const float* __restrict__ b1, const float* __restrict__ b2,
    const float* __restrict__ bg,
    const unsigned short* __restrict__ w1t, const unsigned short* __restrict__ w2t,
    const unsigned short* __restrict__ wgt, const unsigned short* __restrict__ wgb,
    const float* __restrict__ g_out, const float* __restrict__ b_out,
    float* __restrict__ out) {
  __shared__ float fused[4][16][DDIM];  // 32 KB
  const int lane = threadIdx.x & 31;
  const int wv = threadIdx.x >> 5;
  const int m0 = blockIdx.x * 64 + wv * 16;
  const int ml = lane & 15;
  const int hi = lane >> 4;

  Frag16 a1[4], a2[4];
  const unsigned short* xp = xnb + (size_t)(m0 + ml) * DDIM;
  const unsigned short* vp = v2b + (size_t)(m0 + ml) * DDIM;
#pragma unroll
  for (int kc = 0; kc < 4; ++kc) {
    const int base = kc * 32 + hi * 16;
    a1[kc].q[0] = *reinterpret_cast<const uint4*>(xp + base);
    a1[kc].q[1] = *reinterpret_cast<const uint4*>(xp + base + 8);
    a2[kc].q[0] = *reinterpret_cast<const uint4*>(vp + base);
    a2[kc].q[1] = *reinterpret_cast<const uint4*>(vp + base + 8);
  }

#pragma unroll
  for (int t = 0; t < 8; ++t) {
    const int col = t * 16 + ml;
    v8f c1 = (v8f){0.f, 0.f, 0.f, 0.f, 0.f, 0.f, 0.f, 0.f};
    v8f c2 = c1, cg = c1;
#pragma unroll
    for (int kc = 0; kc < 4; ++kc) {
      const int kb = kc * 32 + hi * 16;
      Frag16 bw1, bw2, bgt, bgb;
      bw1.q[0] = *reinterpret_cast<const uint4*>(w1t + col * DDIM + kb);
      bw1.q[1] = *reinterpret_cast<const uint4*>(w1t + col * DDIM + kb + 8);
      bw2.q[0] = *reinterpret_cast<const uint4*>(w2t + col * DDIM + kb);
      bw2.q[1] = *reinterpret_cast<const uint4*>(w2t + col * DDIM + kb + 8);
      bgt.q[0] = *reinterpret_cast<const uint4*>(wgt + col * DDIM + kb);
      bgt.q[1] = *reinterpret_cast<const uint4*>(wgt + col * DDIM + kb + 8);
      bgb.q[0] = *reinterpret_cast<const uint4*>(wgb + col * DDIM + kb);
      bgb.q[1] = *reinterpret_cast<const uint4*>(wgb + col * DDIM + kb + 8);
      c1 = __builtin_amdgcn_wmma_f32_16x16x32_bf16(false, a1[kc].v, false, bw1.v,
                                                   (short)0, c1, false, false);
      c2 = __builtin_amdgcn_wmma_f32_16x16x32_bf16(false, a2[kc].v, false, bw2.v,
                                                   (short)0, c2, false, false);
      cg = __builtin_amdgcn_wmma_f32_16x16x32_bf16(false, a1[kc].v, false, bgt.v,
                                                   (short)0, cg, false, false);
      cg = __builtin_amdgcn_wmma_f32_16x16x32_bf16(false, a2[kc].v, false, bgb.v,
                                                   (short)0, cg, false, false);
    }
    const float bb1 = b1[col], bb2 = b2[col], bbg = bg[col];
#pragma unroll
    for (int r = 0; r < 8; ++r) {
      const int row = m0 + hi * 8 + r;
      const float gv = 1.0f / (1.0f + __expf(-(cg[r] + bbg)));
      float h = gv * (c1[r] + bb1) + (1.0f - gv) * (c2[r] + bb2);
      h += x[(size_t)row * DDIM + col];  // residual
      fused[wv][hi * 8 + r][col] = h;
    }
  }
  __syncthreads();

  // output layernorm: 2 lanes per row, 64 columns each
  const int rl = lane & 15;
  const int half = lane >> 4;
  float s = 0.f;
#pragma unroll 8
  for (int j = 0; j < 64; ++j) s += fused[wv][rl][half * 64 + j];
  s += __shfl_xor(s, 16, 32);
  const float mean = s * (1.0f / DDIM);
  float ss = 0.f;
#pragma unroll 8
  for (int j = 0; j < 64; ++j) {
    const float d = fused[wv][rl][half * 64 + j] - mean;
    ss += d * d;
  }
  ss += __shfl_xor(ss, 16, 32);
  const float inv = rsqrtf(ss * (1.0f / DDIM) + 1e-5f);
  float* op = out + (size_t)(m0 + rl) * DDIM;
#pragma unroll 8
  for (int j = 0; j < 64; ++j) {
    const int col = half * 64 + j;
    op[col] = (fused[wv][rl][col] - mean) * inv * g_out[col] + b_out[col];
  }
}

// ---------------------------------------------------------------------------
extern "C" void kernel_launch(void* const* d_in, const int* in_sizes, int n_in,
                              void* d_out, int out_size, void* d_ws, size_t ws_size,
                              hipStream_t stream) {
  const float* x     = (const float*)d_in[0];
  const float* adj   = (const float*)d_in[1];
  const float* u     = (const float*)d_in[2];
  const float* W1    = (const float*)d_in[3];
  const float* b1    = (const float*)d_in[4];
  const float* W2    = (const float*)d_in[5];
  const float* b2    = (const float*)d_in[6];
  const float* Wg    = (const float*)d_in[7];
  const float* bg    = (const float*)d_in[8];
  const float* Wn    = (const float*)d_in[9];
  const float* bn    = (const float*)d_in[10];
  const float* g_in  = (const float*)d_in[11];
  const float* b_in  = (const float*)d_in[12];
  const float* g_out = (const float*)d_in[13];
  const float* b_out = (const float*)d_in[14];
  float* out = (float*)d_out;

  // workspace carve (~9.6 MB)
  unsigned short* xnb = (unsigned short*)d_ws;                 // [N][128] bf16
  unsigned short* ztb = xnb + (size_t)NROWS * DDIM;            // [128][N] bf16
  unsigned short* v2b = ztb + (size_t)NROWS * DDIM;            // [N][128] bf16
  float* gate = (float*)(v2b + (size_t)NROWS * DDIM);          // [N] f32
  unsigned short* w1t = (unsigned short*)(gate + NROWS);       // [128][128] bf16
  unsigned short* w2t = w1t + DDIM * DDIM;
  unsigned short* wgt = w2t + DDIM * DDIM;
  unsigned short* wgb = wgt + DDIM * DDIM;

  k_prep_w<<<dim3(DDIM, 4), 128, 0, stream>>>(W1, W2, Wg, w1t, w2t, wgt, wgb);
  k_gate_xnorm<<<NROWS / 8, 256, 0, stream>>>(x, u, Wn, bn, g_in, b_in, xnb, ztb, gate);
  k_gemm_big<<<NROWS / 64, 64, 0, stream>>>(adj, ztb, gate, v2b);
  k_epilogue<<<NROWS / 64, 128, 0, stream>>>(xnb, v2b, x, b1, b2, bg,
                                             w1t, w2t, wgt, wgb, g_out, b_out, out);
}
